// HeteroEdgeAwareAttention_39204461478359
// MI455X (gfx1250) — matrix operand order repaired and checked by reading, not compiled, not run on hardware
//
#include <hip/hip_runtime.h>
#include <hip/hip_bf16.h>
#include <math.h>

typedef __attribute__((ext_vector_type(16))) __bf16 v16bf;
typedef __attribute__((ext_vector_type(8)))  __bf16 v8bf;
typedef __attribute__((ext_vector_type(8)))  float  v8f;

// ---------- helpers ----------

__device__ __forceinline__ __bf16 f2bf(float f) {
    unsigned u = __float_as_uint(f);
    u += 0x7fffu + ((u >> 16) & 1u);            // round-to-nearest-even
    unsigned short h = (unsigned short)(u >> 16);
    return __builtin_bit_cast(__bf16, h);
}

__device__ __forceinline__ void atomicMaxF(float* addr, float val) {
    if (val >= 0.0f) atomicMax((int*)addr, __float_as_int(val));
    else             atomicMin((unsigned int*)addr, __float_as_uint(val));
}

__device__ __forceinline__ v16bf pack16(v8bf lo, v8bf hi) {
    v16bf f;
#pragma unroll
    for (int i = 0; i < 8; ++i) { f[i] = lo[i]; f[8 + i] = hi[i]; }
    return f;
}

// ---------- GEMM: C[M x 128] = A[M x K] @ W[K x 128] + bias (+ addsrc) ----------
// 256 threads = 8 waves. Block tile: 32 rows x 128 cols.
// Wave w: row half (w>>2)*16, col pair (w&3)*32 -> two 16x16 WMMA accumulators
// sharing one A fragment. Fragments double-buffered over k to hide DS latency.

template<int K, bool HAS_ADD>
__global__ __launch_bounds__(256)
void gemm_bias_wmma(const float* __restrict__ A, const float* __restrict__ W,
                    const float* __restrict__ bias, const float* __restrict__ addsrc,
                    float* __restrict__ C, int M) {
    constexpr int KP = K + 8;                    // padded row stride (elements)
    __shared__ __bf16 lds_a[32 * KP];
    __shared__ __bf16 lds_w[128 * KP];

    const int tid = threadIdx.x;
    const int bm  = blockIdx.x;

    // stage A tile (32 x K); clamp rows (keeps EXEC full through WMMA)
    for (int i = tid; i < 32 * K; i += 256) {
        int mrow = i / K, kk = i - mrow * K;
        int row  = bm * 32 + mrow;
        if (row >= M) row = M - 1;
        lds_a[mrow * KP + kk] = f2bf(A[(size_t)row * K + kk]);
    }
    // stage W transposed (128 cols x K)
    for (int i = tid; i < 128 * K; i += 256) {
        int kk = i >> 7, n = i & 127;
        lds_w[n * KP + kk] = f2bf(W[(size_t)kk * 128 + n]);
    }
    __syncthreads();

    const int lane  = tid & 31;
    const int wv    = tid >> 5;
    const int half  = lane >> 4;                 // 0: lanes 0-15, 1: lanes 16-31
    const int l15   = lane & 15;
    const int mhalf = wv >> 2;                   // 0/1 -> 16-row slice
    const int npair = wv & 3;                    // 0..3 -> 32-col slice

    // A fragment base: lane row; half 0 holds K {0..7,16..23}, half 1 {8..15,24..31}
    const __bf16* arow  = &lds_a[(mhalf * 16 + l15) * KP + half * 8];
    // B fragment base: lane col; half 0 holds K {0..7,16..23}, half 1 {8..15,24..31}
    const __bf16* brow0 = &lds_w[(npair * 32 + l15) * KP + half * 8];
    const __bf16* brow1 = &lds_w[(npair * 32 + 16 + l15) * KP + half * 8];

    auto lda = [&](int k0) -> v16bf {
        return pack16(*(const v8bf*)(arow + k0), *(const v8bf*)(arow + k0 + 16));
    };
    auto ldb = [&](const __bf16* base, int k0) -> v16bf {
        return pack16(*(const v8bf*)(base + k0), *(const v8bf*)(base + k0 + 16));
    };

    v8f acc0 = {}, acc1 = {};
    v16bf a_c  = lda(0);
    v16bf b0_c = ldb(brow0, 0);
    v16bf b1_c = ldb(brow1, 0);
#pragma unroll
    for (int k0 = 0; k0 < K; k0 += 32) {
        if (k0 + 32 < K) {
            v16bf a_n  = lda(k0 + 32);
            v16bf b0_n = ldb(brow0, k0 + 32);
            v16bf b1_n = ldb(brow1, k0 + 32);
            acc0 = __builtin_amdgcn_wmma_f32_16x16x32_bf16(
                false, a_c, false, b0_c, (short)0, acc0, false, false);
            acc1 = __builtin_amdgcn_wmma_f32_16x16x32_bf16(
                false, a_c, false, b1_c, (short)0, acc1, false, false);
            a_c = a_n; b0_c = b0_n; b1_c = b1_n;
        } else {
            acc0 = __builtin_amdgcn_wmma_f32_16x16x32_bf16(
                false, a_c, false, b0_c, (short)0, acc0, false, false);
            acc1 = __builtin_amdgcn_wmma_f32_16x16x32_bf16(
                false, a_c, false, b1_c, (short)0, acc1, false, false);
        }
    }

    // C layout: VGPR r -> (M = r or 8+r, N = lane&15)
    const int col0 = npair * 32 + l15;
    const int col1 = col0 + 16;
    const float bv0 = bias[col0];
    const float bv1 = bias[col1];
    const int rowbase = bm * 32 + mhalf * 16 + half * 8;

    auto emit = [&](int row, int col, float v) {
        size_t off = (size_t)row * 128 + col;
        float val = v;
        if (HAS_ADD) val += addsrc[off];
        C[off] = val;
    };

    if (bm * 32 + 32 <= M) {                     // uniform full-tile fast path
#pragma unroll
        for (int r = 0; r < 8; ++r) {
            emit(rowbase + r, col0, acc0[r] + bv0);
            emit(rowbase + r, col1, acc1[r] + bv1);
        }
    } else {
#pragma unroll
        for (int r = 0; r < 8; ++r) {
            if (rowbase + r < M) {
                emit(rowbase + r, col0, acc0[r] + bv0);
                emit(rowbase + r, col1, acc1[r] + bv1);
            }
        }
    }
}

// ---------- edge kernels ----------

// one thread per (edge, head): s = scale * dot(q[dst], k[src]+e) ; atomic max into m
__global__ __launch_bounds__(256)
void edge_scores_kernel(const float* __restrict__ q, const float* __restrict__ k,
                        const float* __restrict__ ef,
                        const int* __restrict__ src, const int* __restrict__ dst,
                        float* __restrict__ s, float* __restrict__ m, int E) {
    int idx = blockIdx.x * 256 + threadIdx.x;
    if (idx >= E * 8) return;
    int e = idx >> 3, h = idx & 7;
    int sA = src[e], dA = dst[e];
    const float4* qp = (const float4*)(q  + (size_t)dA * 128 + h * 16);
    const float4* kp = (const float4*)(k  + (size_t)sA * 128 + h * 16);
    const float4* ep = (const float4*)(ef + (size_t)e  * 128 + h * 16);
    float acc = 0.0f;
#pragma unroll
    for (int i = 0; i < 4; ++i) {
        float4 qv = qp[i], kv = kp[i], ev = ep[i];
        acc += qv.x * (kv.x + ev.x) + qv.y * (kv.y + ev.y)
             + qv.z * (kv.z + ev.z) + qv.w * (kv.w + ev.w);
    }
    acc *= 0.25f;                                // D = 16 -> 1/sqrt(16)
    s[idx] = acc;
    atomicMaxF(&m[(size_t)dA * 8 + h], acc);
}

// one thread per (edge, head): ex = exp(s - m[dst]); den[dst] += ex
__global__ __launch_bounds__(256)
void edge_exp_kernel(float* __restrict__ s, const float* __restrict__ m,
                     float* __restrict__ den, const int* __restrict__ dst, int E) {
    int idx = blockIdx.x * 256 + threadIdx.x;
    if (idx >= E * 8) return;
    int e = idx >> 3, h = idx & 7;
    int dA = dst[e];
    float ex = expf(s[idx] - m[(size_t)dA * 8 + h]);
    s[idx] = ex;
    atomicAdd(&den[(size_t)dA * 8 + h], ex);
}

// one thread per (edge, channel): agg[dst,c] += (ex/den) * (v[src,c] + e[c])
__global__ __launch_bounds__(256)
void edge_agg_kernel(const float* __restrict__ s, const float* __restrict__ den,
                     const float* __restrict__ v, const float* __restrict__ ef,
                     const int* __restrict__ src, const int* __restrict__ dst,
                     float* __restrict__ agg, int E) {
    long long idx = (long long)blockIdx.x * 256 + threadIdx.x;
    if (idx >= (long long)E * 128) return;
    int e = (int)(idx >> 7), c = (int)(idx & 127);
    int h = c >> 4;
    int sA = src[e], dA = dst[e];
    float a = s[(size_t)e * 8 + h] / den[(size_t)dA * 8 + h];
    float val = a * (v[(size_t)sA * 128 + c] + ef[(size_t)e * 128 + c]);
    atomicAdd(&agg[(size_t)dA * 128 + c], val);
}

// ---------- misc ----------

__global__ __launch_bounds__(256)
void fill_f32(float* __restrict__ p, float v, long long n) {
    long long i = (long long)blockIdx.x * 256 + threadIdx.x;
    if (i < n) p[i] = v;
}

// one wave per 128-wide row, 4 elems/lane, in-place capable
__global__ __launch_bounds__(256)
void layernorm_kernel(float* __restrict__ x, const float* __restrict__ g,
                      const float* __restrict__ b, int N) {
    int row = blockIdx.x * 8 + (threadIdx.x >> 5);
    if (row >= N) return;
    int lane = threadIdx.x & 31;
    float4 v = *(const float4*)(x + (size_t)row * 128 + lane * 4);
    float sum = v.x + v.y + v.z + v.w;
#pragma unroll
    for (int o = 16; o > 0; o >>= 1) sum += __shfl_xor(sum, o, 32);
    float mu = sum * (1.0f / 128.0f);
    float4 d = make_float4(v.x - mu, v.y - mu, v.z - mu, v.w - mu);
    float sq = d.x * d.x + d.y * d.y + d.z * d.z + d.w * d.w;
#pragma unroll
    for (int o = 16; o > 0; o >>= 1) sq += __shfl_xor(sq, o, 32);
    float rs = rsqrtf(sq * (1.0f / 128.0f) + 1e-5f);
    float4 gg = *(const float4*)(g + lane * 4);
    float4 bb = *(const float4*)(b + lane * 4);
    float4 o4 = make_float4(d.x * rs * gg.x + bb.x, d.y * rs * gg.y + bb.y,
                            d.z * rs * gg.z + bb.z, d.w * rs * gg.w + bb.w);
    *(float4*)(x + (size_t)row * 128 + lane * 4) = o4;
}

// ---------- launch ----------

extern "C" void kernel_launch(void* const* d_in, const int* in_sizes, int n_in,
                              void* d_out, int out_size, void* d_ws, size_t ws_size,
                              hipStream_t stream) {
    const float* x_pc  = (const float*)d_in[0];
    const float* x_gr  = (const float*)d_in[1];
    const float* ea_pp = (const float*)d_in[2];
    const float* ea_pg = (const float*)d_in[3];
    const int*   ei_pp = (const int*)d_in[4];
    const int*   ei_pg = (const int*)d_in[5];

    const int N_pc = in_sizes[0] / 128;
    const int N_gr = in_sizes[1] / 128;
    const int E_pp = in_sizes[2] / 64;
    const int E_pg = in_sizes[3] / 64;

    const float *Wq_pp=(const float*)d_in[6],  *bq_pp=(const float*)d_in[7];
    const float *Wk_pp=(const float*)d_in[8],  *bk_pp=(const float*)d_in[9];
    const float *Wv_pp=(const float*)d_in[10], *bv_pp=(const float*)d_in[11];
    const float *We_pp=(const float*)d_in[12], *be_pp=(const float*)d_in[13];
    const float *Wo_pp=(const float*)d_in[14], *bo_pp=(const float*)d_in[15];
    const float *Wq_pg=(const float*)d_in[16], *bq_pg=(const float*)d_in[17];
    const float *Wk_pg=(const float*)d_in[18], *bk_pg=(const float*)d_in[19];
    const float *Wv_pg=(const float*)d_in[20], *bv_pg=(const float*)d_in[21];
    const float *We_pg=(const float*)d_in[22], *be_pg=(const float*)d_in[23];
    const float *Wo_pg=(const float*)d_in[24], *bo_pg=(const float*)d_in[25];
    const float *Wnp_pc=(const float*)d_in[26], *bnp_pc=(const float*)d_in[27];
    const float *g_pc  =(const float*)d_in[28], *bln_pc=(const float*)d_in[29];
    const float *Wnp_gr=(const float*)d_in[30], *bnp_gr=(const float*)d_in[31];
    const float *g_gr  =(const float*)d_in[32], *bln_gr=(const float*)d_in[33];

    const int *src_pp = ei_pp, *dst_pp = ei_pp + E_pp;
    const int *src_pg = ei_pg, *dst_pg = ei_pg + E_pg;

    float* out_pc = (float*)d_out;
    float* out_gr = out_pc + (size_t)N_pc * 128;

    // workspace carve-up (phased: pg phase reuses q/k/v/e/s/m/den buffers)
    float* ws = (float*)d_ws;
    size_t o = 0;
    float* bufQ  = ws + o; o += (size_t)N_pc * 128;
    float* bufK  = ws + o; o += (size_t)N_pc * 128;
    float* bufV  = ws + o; o += (size_t)N_pc * 128;
    float* bufE  = ws + o; o += (size_t)E_pp * 128;
    float* bufS  = ws + o; o += (size_t)E_pp * 8;
    float* bufM  = ws + o; o += (size_t)N_pc * 8;
    float* bufD  = ws + o; o += (size_t)N_pc * 8;
    float* aggPC = ws + o; o += (size_t)N_pc * 128;
    float* aggGR = ws + o; o += (size_t)N_gr * 128;
    float* npPC  = ws + o; o += (size_t)N_pc * 128;
    float* npGR  = ws + o; o += (size_t)N_gr * 128;

    auto gemm = [&](const float* A, const float* W, const float* b,
                    const float* add, float* C, int M, int K) {
        dim3 grid((M + 31) / 32);
        if (K == 128) {
            if (add) hipLaunchKernelGGL((gemm_bias_wmma<128, true>),  grid, dim3(256), 0, stream, A, W, b, add, C, M);
            else     hipLaunchKernelGGL((gemm_bias_wmma<128, false>), grid, dim3(256), 0, stream, A, W, b, nullptr, C, M);
        } else {
            if (add) hipLaunchKernelGGL((gemm_bias_wmma<64, true>),   grid, dim3(256), 0, stream, A, W, b, add, C, M);
            else     hipLaunchKernelGGL((gemm_bias_wmma<64, false>),  grid, dim3(256), 0, stream, A, W, b, nullptr, C, M);
        }
    };
    auto fill = [&](float* p, float v, long long n) {
        hipLaunchKernelGGL(fill_f32, dim3((unsigned)((n + 255) / 256)), dim3(256), 0, stream, p, v, n);
    };

    // node projections
    gemm(x_pc, Wnp_pc, bnp_pc, nullptr, npPC, N_pc, 128);
    gemm(x_gr, Wnp_gr, bnp_gr, nullptr, npGR, N_gr, 128);

    // ---- pp edge type (pc -> pc) ----
    gemm(x_pc,  Wq_pp, bq_pp, nullptr, bufQ, N_pc, 128);
    gemm(x_pc,  Wk_pp, bk_pp, nullptr, bufK, N_pc, 128);
    gemm(x_pc,  Wv_pp, bv_pp, nullptr, bufV, N_pc, 128);
    gemm(ea_pp, We_pp, be_pp, nullptr, bufE, E_pp, 64);

    fill(bufM, -INFINITY, (long long)N_pc * 8);
    fill(bufD, 0.0f,      (long long)N_pc * 8);
    fill(aggPC, 0.0f,     (long long)N_pc * 128);

    hipLaunchKernelGGL(edge_scores_kernel, dim3((E_pp * 8 + 255) / 256), dim3(256), 0, stream,
                       bufQ, bufK, bufE, src_pp, dst_pp, bufS, bufM, E_pp);
    hipLaunchKernelGGL(edge_exp_kernel, dim3((E_pp * 8 + 255) / 256), dim3(256), 0, stream,
                       bufS, bufM, bufD, dst_pp, E_pp);
    hipLaunchKernelGGL(edge_agg_kernel,
                       dim3((unsigned)(((long long)E_pp * 128 + 255) / 256)), dim3(256), 0, stream,
                       bufS, bufD, bufV, bufE, src_pp, dst_pp, aggPC, E_pp);
    gemm(aggPC, Wo_pp, bo_pp, npPC, out_pc, N_pc, 128);

    // ---- pg edge type (pc -> gr) ----
    gemm(x_gr,  Wq_pg, bq_pg, nullptr, bufQ, N_gr, 128);
    gemm(x_pc,  Wk_pg, bk_pg, nullptr, bufK, N_pc, 128);
    gemm(x_pc,  Wv_pg, bv_pg, nullptr, bufV, N_pc, 128);
    gemm(ea_pg, We_pg, be_pg, nullptr, bufE, E_pg, 64);

    fill(bufM, -INFINITY, (long long)N_gr * 8);
    fill(bufD, 0.0f,      (long long)N_gr * 8);
    fill(aggGR, 0.0f,     (long long)N_gr * 128);

    hipLaunchKernelGGL(edge_scores_kernel, dim3((E_pg * 8 + 255) / 256), dim3(256), 0, stream,
                       bufQ, bufK, bufE, src_pg, dst_pg, bufS, bufM, E_pg);
    hipLaunchKernelGGL(edge_exp_kernel, dim3((E_pg * 8 + 255) / 256), dim3(256), 0, stream,
                       bufS, bufM, bufD, dst_pg, E_pg);
    hipLaunchKernelGGL(edge_agg_kernel,
                       dim3((unsigned)(((long long)E_pg * 128 + 255) / 256)), dim3(256), 0, stream,
                       bufS, bufD, bufV, bufE, src_pg, dst_pg, aggGR, E_pg);
    gemm(aggGR, Wo_pg, bo_pg, npGR, out_gr, N_gr, 128);

    // ---- layernorms (in place on d_out) ----
    hipLaunchKernelGGL(layernorm_kernel, dim3((N_pc + 7) / 8), dim3(256), 0, stream,
                       out_pc, g_pc, bln_pc, N_pc);
    hipLaunchKernelGGL(layernorm_kernel, dim3((N_gr + 7) / 8), dim3(256), 0, stream,
                       out_gr, g_gr, bln_gr, N_gr);
}
